// GraphTransformerEncoder_5738076307740
// MI455X (gfx1250) — matrix operand order repaired and checked by reading, not compiled, not run on hardware
//
#include <hip/hip_runtime.h>

// ---------------------------------------------------------------------------
// GraphGPS encoder for MI455X (gfx1250): all dense GEMMs via
// v_wmma_f32_16x16x32_f16 (f16 operands, f32 accum) with LDS double-buffered
// tiles; scatter GAT via f32 atomics; per-(graph,head) attention with
// LDS-resident score/prob tiles and LDS-staged transposed V;
// global_prefetch_b8 one tile ahead.
// ---------------------------------------------------------------------------

typedef __attribute__((ext_vector_type(16))) _Float16 v16h;
typedef __attribute__((ext_vector_type(8)))  _Float16 v8h;
typedef __attribute__((ext_vector_type(4)))  _Float16 v4h;
typedef __attribute__((ext_vector_type(8)))  float    v8f;

#define WMMA_F16(a, b, c) \
  __builtin_amdgcn_wmma_f32_16x16x32_f16(false, (a), false, (b), (short)0, (c), false, false)

__device__ __forceinline__ v16h cat8(v8h a, v8h b) {
  return __builtin_shufflevector(a, b, 0,1,2,3,4,5,6,7,8,9,10,11,12,13,14,15);
}

// convert 8 consecutive f32 (16B-aligned) to f16, scaled
__device__ __forceinline__ v8h cvt8(const float* __restrict__ p, float s) {
  const float4 a = *(const float4*)p;
  const float4 b = *(const float4*)(p + 4);
  v8h r = {(_Float16)(a.x * s), (_Float16)(a.y * s), (_Float16)(a.z * s), (_Float16)(a.w * s),
           (_Float16)(b.x * s), (_Float16)(b.y * s), (_Float16)(b.z * s), (_Float16)(b.w * s)};
  return r;
}

__device__ __forceinline__ v4h cvt4(float4 f) {
  v4h r = {(_Float16)f.x, (_Float16)f.y, (_Float16)f.z, (_Float16)f.w};
  return r;
}

// ---------------------------------------------------------------------------
// Generic WMMA GEMM: C[M,Ncol] = act(A[M,K] @ W[Ncol,K]^T + bias)
// block tile 128x64, 8 waves (4x2), wave tile 32x32 (2x2 WMMA frags), K step 32
// LDS double-buffered: global loads for tile k+1 issued before WMMAs of tile k.
// M%128==0, Ncol%64==0, K%32==0
// ---------------------------------------------------------------------------
__global__ void __launch_bounds__(256)
gemm_wmma(const float* __restrict__ A, const float* __restrict__ W,
          const float* __restrict__ bias, float* __restrict__ C,
          int M, int Ncol, int K, int relu) {
  __shared__ _Float16 As[2][128 * 32];
  __shared__ _Float16 Ws[2][64 * 32];
  const int tid = threadIdx.x;
  const int wave = tid >> 5, lane = tid & 31;
  const int L = lane & 15, hi = lane >> 4;
  const int row0 = (wave >> 1) * 32, col0 = (wave & 1) * 32;
  const long long bm = (long long)blockIdx.x * 128;
  const long long bn = (long long)blockIdx.y * 64;

  float4 fa[4], fw[2];
  // ---- prologue: stage tile k0=0 into buffer 0 (loads batched, then stores)
  for (int j = 0; j < 4; ++j) {
    const int s = tid + 256 * j;
    fa[j] = *(const float4*)(A + (bm + (s >> 3)) * (long long)K + (s & 7) * 4);
  }
  for (int j = 0; j < 2; ++j) {
    const int s = tid + 256 * j;
    fw[j] = *(const float4*)(W + (bn + (s >> 3)) * (long long)K + (s & 7) * 4);
  }
  for (int j = 0; j < 4; ++j) {
    const int s = tid + 256 * j;
    *(v4h*)(&As[0][(s >> 3) * 32 + (s & 7) * 4]) = cvt4(fa[j]);
  }
  for (int j = 0; j < 2; ++j) {
    const int s = tid + 256 * j;
    *(v4h*)(&Ws[0][(s >> 3) * 32 + (s & 7) * 4]) = cvt4(fw[j]);
  }
  __syncthreads();

  v8f acc[2][2] = {};
  for (int k0 = 0; k0 < K; k0 += 32) {
    const int cur = (k0 >> 5) & 1;
    const bool more = (k0 + 32) < K;
    // ---- kick off next tile's global loads before computing
    if (more) {
      for (int j = 0; j < 4; ++j) {
        const int s = tid + 256 * j;
        fa[j] = *(const float4*)(A + (bm + (s >> 3)) * (long long)K + k0 + 32 + (s & 7) * 4);
      }
      for (int j = 0; j < 2; ++j) {
        const int s = tid + 256 * j;
        fw[j] = *(const float4*)(W + (bn + (s >> 3)) * (long long)K + k0 + 32 + (s & 7) * 4);
      }
      if (k0 + 64 < K) {  // pull tile k+2 toward the WGP (global_prefetch_b8)
        __builtin_prefetch(A + (bm + (tid >> 3)) * (long long)K + k0 + 64 + (tid & 7) * 4, 0, 3);
        __builtin_prefetch(W + (bn + (tid & 63)) * (long long)K + k0 + 64, 0, 3);
      }
    }
    // ---- compute from current LDS buffer
    v16h af[2], bf[2];
    for (int i = 0; i < 2; ++i) {
      // A layout: lanes 0-15 hold K{0..7,16..23}, lanes 16-31 K{8..15,24..31}
      const _Float16* p = &As[cur][(row0 + 16 * i + L) * 32 + hi * 8];
      af[i] = cat8(*(const v8h*)p, *(const v8h*)(p + 16));
    }
    for (int j = 0; j < 2; ++j) {
      // B layout: lane n = column n; lanes 0-15 K{0..7,16..23}, lanes 16-31 K{8..15,24..31}
      const _Float16* p = &Ws[cur][(col0 + 16 * j + L) * 32 + hi * 8];
      bf[j] = cat8(*(const v8h*)p, *(const v8h*)(p + 16));
    }
    for (int i = 0; i < 2; ++i)
      for (int j = 0; j < 2; ++j)
        acc[i][j] = WMMA_F16(af[i], bf[j], acc[i][j]);
    // ---- convert + store next tile, single barrier per K-step
    if (more) {
      const int nxt = cur ^ 1;
      for (int j = 0; j < 4; ++j) {
        const int s = tid + 256 * j;
        *(v4h*)(&As[nxt][(s >> 3) * 32 + (s & 7) * 4]) = cvt4(fa[j]);
      }
      for (int j = 0; j < 2; ++j) {
        const int s = tid + 256 * j;
        *(v4h*)(&Ws[nxt][(s >> 3) * 32 + (s & 7) * 4]) = cvt4(fw[j]);
      }
      __syncthreads();
    }
  }
  // epilogue: C/D layout VGPR r -> M = r (lanes 0-15) / 8+r (lanes 16-31), N = lane%16
  for (int i = 0; i < 2; ++i)
    for (int j = 0; j < 2; ++j) {
      long long n = bn + col0 + 16 * j + L;
      float bv = bias ? bias[n] : 0.f;
      for (int r = 0; r < 8; ++r) {
        long long m = bm + row0 + 16 * i + hi * 8 + r;
        float v = acc[i][j][r] + bv;
        if (relu) v = v > 0.f ? v : 0.f;
        C[m * Ncol + n] = v;
      }
    }
}

// ---------------------------------------------------------------------------
// MHA per (graph, head, 32-row query block). qkv: [N,768] (q|k|v).
// out: [N,256] attention output (pre out-proj).
// LDS plan (one 64KB buffer):
//   phase 1: f32 scores S[32][512] (all 64KB)
//   phase 2: probs f16 P[32][512] alias the first 32KB
//   phase 3: transposed V tile Vt[64][256] f16 in the upper 32KB (2 k-halves)
// ---------------------------------------------------------------------------
__global__ void __launch_bounds__(256)
attn_kernel(const float* __restrict__ qkv, float* __restrict__ out) {
  __shared__ float Ssc[32 * 512];  // 64KB
  const int tid = threadIdx.x;
  const int wave = tid >> 5, lane = tid & 31;
  const int L = lane & 15, hi = lane >> 4;
  const int blk = blockIdx.x;
  const int qb = blk & 15, h = (blk >> 4) & 3, b = blk >> 6;
  const int q0 = qb * 32;
  const long long base = (long long)b * 512;
  const float scale = 0.125f;  // 1/sqrt(64)

  // ---- phase 1: S[32,512] = (Q*scale) @ K^T ; wave w -> columns [64w,64w+64)
  {
    v8f acc[2][4] = {};
    for (int ks = 0; ks < 2; ++ks) {  // hd = 64 -> 2 K-steps of 32
      const int k0 = ks * 32;
      v16h af[2], bf[4];
      for (int i = 0; i < 2; ++i) {
        const float* p = qkv + (base + q0 + 16 * i + L) * 768 + h * 64 + k0 + hi * 8;
        af[i] = cat8(cvt8(p, scale), cvt8(p + 16, scale));
      }
      for (int j = 0; j < 4; ++j) {
        const long long key = base + 64 * wave + 16 * j + L;
        const float* p = qkv + key * 768 + 256 + h * 64 + k0 + hi * 8;
        bf[j] = cat8(cvt8(p, 1.f), cvt8(p + 16, 1.f));
      }
      for (int i = 0; i < 2; ++i)
        for (int j = 0; j < 4; ++j)
          acc[i][j] = WMMA_F16(af[i], bf[j], acc[i][j]);
    }
    for (int i = 0; i < 2; ++i)
      for (int j = 0; j < 4; ++j) {
        int col = 64 * wave + 16 * j + L;
        for (int r = 0; r < 8; ++r)
          Ssc[(16 * i + hi * 8 + r) * 512 + col] = acc[i][j][r];
      }
  }
  __syncthreads();

  // ---- phase 2: row softmax; wave handles 4 rows, lane handles 16 cols
  {
    float vals[4][16];
    for (int rr = 0; rr < 4; ++rr) {
      const float* row = Ssc + (wave * 4 + rr) * 512 + lane * 16;
      for (int c = 0; c < 16; ++c) vals[rr][c] = row[c];
    }
    __syncthreads();  // scores consumed; safe to alias LDS as f16 probs
    _Float16* Ph = (_Float16*)Ssc;
    for (int rr = 0; rr < 4; ++rr) {
      float mx = -3.0e38f;
      for (int c = 0; c < 16; ++c) mx = fmaxf(mx, vals[rr][c]);
      for (int o = 16; o > 0; o >>= 1) mx = fmaxf(mx, __shfl_xor(mx, o, 32));
      float sum = 0.f;
      for (int c = 0; c < 16; ++c) { vals[rr][c] = __expf(vals[rr][c] - mx); sum += vals[rr][c]; }
      for (int o = 16; o > 0; o >>= 1) sum += __shfl_xor(sum, o, 32);
      const float inv = 1.f / sum;
      _Float16* pr = Ph + (wave * 4 + rr) * 512 + lane * 16;
      for (int c = 0; c < 16; ++c) pr[c] = (_Float16)(vals[rr][c] * inv);
    }
  }
  __syncthreads();

  // ---- phase 3: O[32,64] = P @ V ; wave -> one 16x16 tile.
  // V staged transposed in the upper 32KB of LDS, 256 keys per pass.
  {
    _Float16* Ph = (_Float16*)Ssc;                 // P[32][512], first 32KB
    _Float16* Vt = (_Float16*)Ssc + 32 * 512;      // Vt[64][256], upper 32KB
    const int ti = wave >> 2, tj = wave & 3;
    v8f acc = {};
    for (int halfk = 0; halfk < 2; ++halfk) {
      const int kbase = halfk * 256;
      if (halfk) __syncthreads();  // all waves done with previous Vt
      // stage 256 keys x 64 dims: 4096 float4, 16/thread, 4 loads in flight
      for (int jj = 0; jj < 16; jj += 4) {
        float4 f[4];
        int key[4], c4[4];
        for (int u = 0; u < 4; ++u) {
          const int s = tid + 256 * (jj + u);
          key[u] = s >> 4; c4[u] = s & 15;
          f[u] = *(const float4*)(qkv + (base + kbase + key[u]) * 768 + 512 + h * 64 + c4[u] * 4);
        }
        for (int u = 0; u < 4; ++u) {
          Vt[(c4[u] * 4 + 0) * 256 + key[u]] = (_Float16)f[u].x;
          Vt[(c4[u] * 4 + 1) * 256 + key[u]] = (_Float16)f[u].y;
          Vt[(c4[u] * 4 + 2) * 256 + key[u]] = (_Float16)f[u].z;
          Vt[(c4[u] * 4 + 3) * 256 + key[u]] = (_Float16)f[u].w;
        }
      }
      __syncthreads();
      for (int k0 = 0; k0 < 256; k0 += 32) {
        const _Float16* pa = Ph + (16 * ti + L) * 512 + kbase + k0 + hi * 8;
        v16h af = cat8(*(const v8h*)pa, *(const v8h*)(pa + 16));
        const _Float16* pb = Vt + (16 * tj + L) * 256 + k0 + hi * 8;
        v16h bf = cat8(*(const v8h*)pb, *(const v8h*)(pb + 16));
        acc = WMMA_F16(af, bf, acc);
      }
    }
    for (int r = 0; r < 8; ++r) {
      const long long node = base + q0 + 16 * ti + hi * 8 + r;
      out[node * 256 + h * 64 + 16 * tj + L] = acc[r];
    }
  }
}

// ---------------------------------------------------------------------------
// GAT pieces
// ---------------------------------------------------------------------------
__global__ void gat_dots(const float* __restrict__ xs, const float* __restrict__ a_s,
                         const float* __restrict__ a_d, float* __restrict__ as_o,
                         float* __restrict__ ad_o) {
  const int wave = threadIdx.x >> 5, lane = threadIdx.x & 31;
  const long long n = (long long)blockIdx.x * 8 + wave;
  const float* row = xs + n * 256;
  float s1 = 0.f, s2 = 0.f;
  for (int c = lane; c < 256; c += 32) { const float v = row[c]; s1 += v * a_s[c]; s2 += v * a_d[c]; }
  for (int o = 16; o > 0; o >>= 1) { s1 += __shfl_xor(s1, o, 32); s2 += __shfl_xor(s2, o, 32); }
  if (lane == 0) { as_o[n] = s1; ad_o[n] = s2; }
}

__device__ __forceinline__ void edge_sd(const int* ei, int e, int E, int* s, int* d) {
  if (e < E) { *s = ei[e]; *d = ei[E + e]; } else { *s = *d = e - E; }
}

__global__ void gat_edge1(const int* __restrict__ ei, const float* __restrict__ asv,
                          const float* __restrict__ adv, float* __restrict__ alpha,
                          unsigned* __restrict__ mxenc, int E, int tot) {
  const int e = blockIdx.x * 256 + threadIdx.x;
  if (e >= tot) return;
  int s, d; edge_sd(ei, e, E, &s, &d);
  float a = asv[s] + adv[d];
  a = a > 0.f ? a : 0.2f * a;            // leaky_relu 0.2
  alpha[e] = a;
  unsigned u = __float_as_uint(a);
  u = (u & 0x80000000u) ? ~u : (u | 0x80000000u);  // order-preserving encode
  atomicMax(mxenc + d, u);
}

__global__ void gat_edge2(const int* __restrict__ ei, const float* __restrict__ alpha,
                          const unsigned* __restrict__ mxenc, float* __restrict__ ex,
                          float* __restrict__ den, int E, int tot) {
  const int e = blockIdx.x * 256 + threadIdx.x;
  if (e >= tot) return;
  int s, d; edge_sd(ei, e, E, &s, &d); (void)s;
  const unsigned u = mxenc[d];
  const float mx = __uint_as_float((u & 0x80000000u) ? (u ^ 0x80000000u) : ~u);
  const float x = __expf(alpha[e] - mx);
  ex[e] = x;
  atomicAdd(den + d, x);
}

// one block per edge; thread = channel
__global__ void gat_edge3(const int* __restrict__ ei, const float* __restrict__ ex,
                          const float* __restrict__ den, const float* __restrict__ xs,
                          float* __restrict__ outb, int E) {
  const int e = blockIdx.x;
  int s, d; edge_sd(ei, e, E, &s, &d);
  const float w = ex[e] / den[d];
  const int c = threadIdx.x;
  atomicAdd(outb + (long long)d * 256 + c, xs[(long long)s * 256 + c] * w);
}

// ---------------------------------------------------------------------------
// BatchNorm (training stats over nodes), LayerNorm, util
// ---------------------------------------------------------------------------
__global__ void bn_stats(const float* __restrict__ a, const float* __restrict__ b,
                         const float* __restrict__ cb, float* __restrict__ sum,
                         float* __restrict__ sumsq, int rows_per_block) {
  const int c = threadIdx.x;
  const long long r0 = (long long)blockIdx.x * rows_per_block;
  float s = 0.f, q = 0.f;
  for (int r = 0; r < rows_per_block; ++r) {
    const long long idx = (r0 + r) * 256 + c;
    float v = a[idx];
    if (b) v += b[idx];
    if (cb) v += cb[c];
    s += v; q += v * v;
  }
  atomicAdd(sum + c, s);
  atomicAdd(sumsq + c, q);
}

__global__ void bn_final(const float* __restrict__ sum, const float* __restrict__ sumsq,
                         float* __restrict__ mean, float* __restrict__ rvar, float invN) {
  const int c = threadIdx.x;
  const float m = sum[c] * invN;
  const float v = sumsq[c] * invN - m * m;
  mean[c] = m;
  rvar[c] = rsqrtf(v + 1e-5f);
}

__global__ void bn_apply(const float* __restrict__ a, const float* __restrict__ b,
                         const float* __restrict__ cb, const float* __restrict__ mean,
                         const float* __restrict__ rvar, const float* __restrict__ g,
                         const float* __restrict__ beta, float* __restrict__ out) {
  const long long i = (long long)blockIdx.x * 256 + threadIdx.x;
  const int c = (int)(i & 255);
  float v = a[i];
  if (b) v += b[i];
  if (cb) v += cb[c];
  out[i] = (v - mean[c]) * rvar[c] * g[c] + beta[c];
}

__global__ void layernorm_k(const float* __restrict__ x, const float* __restrict__ g,
                            const float* __restrict__ b, float* __restrict__ o) {
  const int wave = threadIdx.x >> 5, lane = threadIdx.x & 31;
  const long long n = (long long)blockIdx.x * 8 + wave;
  const float* row = x + n * 256;
  float vals[8];
  float s = 0.f;
  for (int j = 0; j < 8; ++j) { vals[j] = row[lane + 32 * j]; s += vals[j]; }
  for (int o2 = 16; o2 > 0; o2 >>= 1) s += __shfl_xor(s, o2, 32);
  const float m = s * (1.f / 256.f);
  float q = 0.f;
  for (int j = 0; j < 8; ++j) { const float d = vals[j] - m; q += d * d; }
  for (int o2 = 16; o2 > 0; o2 >>= 1) q += __shfl_xor(q, o2, 32);
  const float rv = rsqrtf(q * (1.f / 256.f) + 1e-5f);
  for (int j = 0; j < 8; ++j) {
    const int c = lane + 32 * j;
    o[n * 256 + c] = (vals[j] - m) * rv * g[c] + b[c];
  }
}

__global__ void add2_k(const float* __restrict__ a, const float* __restrict__ b,
                       float* __restrict__ o, long long n) {
  const long long i = (long long)blockIdx.x * 256 + threadIdx.x;
  if (i < n) o[i] = a[i] + b[i];
}

__global__ void fill_zero_f(float* p, long long n) {
  const long long i = (long long)blockIdx.x * 256 + threadIdx.x;
  if (i < n) p[i] = 0.f;
}
__global__ void fill_zero_u(unsigned* p, long long n) {
  const long long i = (long long)blockIdx.x * 256 + threadIdx.x;
  if (i < n) p[i] = 0u;  // encoded -inf lower bound for float-max
}

// ---------------------------------------------------------------------------
extern "C" void kernel_launch(void* const* d_in, const int* in_sizes, int n_in,
                              void* d_out, int out_size, void* d_ws, size_t ws_size,
                              hipStream_t stream) {
  (void)in_sizes; (void)n_in; (void)out_size; (void)ws_size;
  constexpr long long NB = 128LL * 512;       // 65536 nodes
  constexpr int HID = 256;
  constexpr int EED = (int)(NB * 8);          // 524288 edges
  constexpr int ETOT = EED + (int)NB;         // + self loops

  const float* x         = (const float*)d_in[0];
  const int*   ei        = (const int*)d_in[1];
  const float* W_in      = (const float*)d_in[2];
  const float* b_in      = (const float*)d_in[3];
  const float* gat_w     = (const float*)d_in[4];
  const float* gat_as    = (const float*)d_in[5];
  const float* gat_ad    = (const float*)d_in[6];
  const float* gat_b     = (const float*)d_in[7];
  const float* attn_in_w = (const float*)d_in[8];
  const float* attn_in_b = (const float*)d_in[9];
  const float* attn_ow   = (const float*)d_in[10];
  const float* attn_ob   = (const float*)d_in[11];
  const float* mlp_w1    = (const float*)d_in[12];
  const float* mlp_b1    = (const float*)d_in[13];
  const float* mlp_w2    = (const float*)d_in[14];
  const float* mlp_b2    = (const float*)d_in[15];
  const float* bn1_g = (const float*)d_in[16];
  const float* bn1_b = (const float*)d_in[17];
  const float* bn2_g = (const float*)d_in[18];
  const float* bn2_b = (const float*)d_in[19];
  const float* bn3_g = (const float*)d_in[20];
  const float* bn3_b = (const float*)d_in[21];
  const float* ln_g  = (const float*)d_in[22];
  const float* ln_b  = (const float*)d_in[23];
  const float* W_out = (const float*)d_in[24];
  const float* b_out = (const float*)d_in[25];

  // ---- workspace carve (floats) ----
  float* ws   = (float*)d_ws;
  float* h    = ws;                     // [NB,256]
  float* bufA = h + NB * HID;           // xs / attn_o
  float* bufB = bufA + NB * HID;        // hl_pre / hl / comb
  float* bufD = bufB + NB * HID;        // hg
  float* qkv  = bufD + NB * HID;        // [NB,768]; later: mid [NB,512] + mout [NB,256]
  float* mid  = qkv;
  float* mout = qkv + NB * 512;
  float* asv  = qkv + NB * 768;
  float* adv  = asv + NB;
  float* alp  = adv + NB;               // [ETOT]
  float* exb  = alp + ETOT;             // [ETOT]
  float* den  = exb + ETOT;             // [NB]
  unsigned* mxenc = (unsigned*)(den + NB);
  float* bnsum  = (float*)(mxenc + NB); // 256
  float* bnsq   = bnsum + 256;
  float* bnmean = bnsq + 256;
  float* bnrvar = bnmean + 256;

  auto gemm = [&](const float* A, const float* W, const float* bias, float* C,
                  int M, int Ncol, int K, int relu) {
    dim3 g(M / 128, Ncol / 64);
    gemm_wmma<<<g, 256, 0, stream>>>(A, W, bias, C, M, Ncol, K, relu);
  };
  auto zf = [&](float* p, long long n) {
    fill_zero_f<<<(int)((n + 255) / 256), 256, 0, stream>>>(p, n);
  };
  auto bn = [&](const float* a, const float* b, const float* cb,
                const float* g, const float* beta, float* out) {
    zf(bnsum, 256); zf(bnsq, 256);
    bn_stats<<<256, 256, 0, stream>>>(a, b, cb, bnsum, bnsq, (int)(NB / 256));
    bn_final<<<1, 256, 0, stream>>>(bnsum, bnsq, bnmean, bnrvar, 1.f / (float)NB);
    bn_apply<<<(int)NB, 256, 0, stream>>>(a, b, cb, bnmean, bnrvar, g, beta, out);
  };

  // h = relu(x @ W_in^T + b_in)    [NB,256], K=128
  gemm(x, W_in, b_in, h, (int)NB, 256, 128, 1);

  for (int i = 0; i < 2; ++i) {
    // ---- local GAT branch ----
    gemm(h, gat_w + (long long)i * 256 * 256, nullptr, bufA, (int)NB, 256, 256, 0);  // xs
    gat_dots<<<(int)(NB / 8), 256, 0, stream>>>(bufA, gat_as + i * 256, gat_ad + i * 256, asv, adv);
    zf(den, NB); zf(bufB, NB * HID);
    fill_zero_u<<<(int)((NB + 255) / 256), 256, 0, stream>>>(mxenc, NB);
    gat_edge1<<<(ETOT + 255) / 256, 256, 0, stream>>>(ei, asv, adv, alp, mxenc, EED, ETOT);
    gat_edge2<<<(ETOT + 255) / 256, 256, 0, stream>>>(ei, alp, mxenc, exb, den, EED, ETOT);
    gat_edge3<<<ETOT, 256, 0, stream>>>(ei, exb, den, bufA, bufB, EED);
    // hl = BN1(agg + gat_b + h)
    bn(bufB, h, gat_b + i * 256, bn1_g + i * 256, bn1_b + i * 256, bufB);

    // ---- global attention branch ----
    gemm(h, attn_in_w + (long long)i * 768 * 256, attn_in_b + i * 768, qkv, (int)NB, 768, 256, 0);
    attn_kernel<<<128 * 4 * 16, 256, 0, stream>>>(qkv, bufA);
    gemm(bufA, attn_ow + (long long)i * 256 * 256, attn_ob + i * 256, bufD, (int)NB, 256, 256, 0);
    // hg = BN2(attn + h)
    bn(bufD, h, nullptr, bn2_g + i * 256, bn2_b + i * 256, bufD);

    // ---- combine + FFN ----
    add2_k<<<(int)NB, 256, 0, stream>>>(bufB, bufD, bufB, NB * HID);  // comb = hl + hg
    gemm(bufB, mlp_w1 + (long long)i * 512 * 256, mlp_b1 + i * 512, mid, (int)NB, 512, 256, 1);
    gemm(mid, mlp_w2 + (long long)i * 256 * 512, mlp_b2 + i * 256, mout, (int)NB, 256, 512, 0);
    bn(bufB, mout, nullptr, bn3_g + i * 256, bn3_b + i * 256, bufB);  // BN3(comb + m)
    layernorm_k<<<(int)(NB / 8), 256, 0, stream>>>(bufB, ln_g + i * 256, ln_b + i * 256, h);
  }

  // out = h @ W_out^T + b_out   [NB,384]
  gemm(h, W_out, b_out, (float*)d_out, (int)NB, 384, 256, 0);
}
